// IBQ_8933531975956
// MI455X (gfx1250) — matrix operand order repaired and checked by reading, not compiled, not run on hardware
//
#include <hip/hip_runtime.h>
#include <hip/hip_bf16.h>

// ---------------------------------------------------------------------------
// Problem constants
// ---------------------------------------------------------------------------
#define N_ROWS   8192      // 8 * 32 * 32 z-vectors
#define E_CODES  16384     // codebook entries
#define CDIM     256       // embedding dim
#define HW       1024      // 32*32

// GEMM tiling
#define ROWS_PER_WG     128    // 8 waves * 16 rows
#define COL_CHUNKS      4
#define COLS_PER_CHUNK  (E_CODES / COL_CHUNKS)   // 4096
#define TILE_COLS       64
#define TILES_PER_CHUNK (COLS_PER_CHUNK / TILE_COLS) // 64
#define TILE_BYTES      (TILE_COLS * CDIM * 2)       // 32 KB, contiguous in enb

// ---------------------------------------------------------------------------
// CDNA5 types
// ---------------------------------------------------------------------------
typedef __attribute__((ext_vector_type(16))) __bf16 bf16x16;
typedef __attribute__((ext_vector_type(8)))  float  f32x8;
typedef __attribute__((ext_vector_type(4)))  unsigned int u32x4;
typedef __attribute__((ext_vector_type(8)))  int    i32x8;
typedef __attribute__((ext_vector_type(4)))  int    i32x4;

union ABPack { u32x4 q[2]; bf16x16 v; };   // 32B: one WMMA A/B operand per lane

#if defined(__has_builtin)
# if __has_builtin(__builtin_amdgcn_tensor_load_to_lds)
#  define VQ_HAVE_TDM 1
# endif
#endif

#ifdef VQ_HAVE_TDM
// Issue a 1-D Tensor-Data-Mover load of one contiguous 32 KB B tile into LDS.
// D# per CDNA5 ISA: group0 = {count/lds_addr/global_addr/type},
// group1 = {mask=0, data_size=8B, tensor_dim0=tile_dim0=4096 elems, 1-D}.
__device__ __forceinline__ void tdm_load_tile(unsigned lds_off, const void* gptr) {
  const unsigned long long ga = (unsigned long long)(uintptr_t)gptr;
  u32x4 g0;
  g0[0] = 1u;                                                // count=1, user mode
  g0[1] = lds_off;                                           // lds_addr (bytes)
  g0[2] = (unsigned)ga;                                      // global_addr[31:0]
  g0[3] = (unsigned)((ga >> 32) & 0x01FFFFFFu) | (2u << 30); // addr[56:32] | type=2
  i32x8 g1;
  g1[0] = (3 << 16);                 // workgroup_mask=0, data_size=3 (8 bytes)
  g1[1] = (int)(4096u << 16);        // tensor_dim0[15:0]=4096 -> bits[63:48]
  g1[2] = (int)(1u << 16);           // tensor_dim0 hi=0; tensor_dim1=1 (unused)
  g1[3] = (int)(4096u << 16);        // tile_dim0=4096 -> bits[127:112]
  g1[4] = 0; g1[5] = 0; g1[6] = 0; g1[7] = 0;
  const i32x4 z4 = {0, 0, 0, 0};
# if __clang_major__ >= 23
  const i32x8 z8 = {0, 0, 0, 0, 0, 0, 0, 0};
  __builtin_amdgcn_tensor_load_to_lds(g0, g1, z4, z4, z8, 0);
# else
  __builtin_amdgcn_tensor_load_to_lds(g0, g1, z4, z4, 0);
# endif
}
#endif

// ---------------------------------------------------------------------------
// Kernel 1: normalize z rows (NCHW strided) -> bf16 zn [N_ROWS][CDIM] + 1/||z||
// ---------------------------------------------------------------------------
__global__ void __launch_bounds__(256)
vq_prep_z(const float* __restrict__ z,
          __hip_bfloat16* __restrict__ znb,
          float* __restrict__ invn) {
  const int n = blockIdx.x;          // 0..8191
  const int c = threadIdx.x;         // 0..255
  const int b = n >> 10, hw = n & (HW - 1);
  const float v = z[(size_t)b * (CDIM * HW) + (size_t)c * HW + hw];

  __shared__ float red[256];
  __shared__ float sinv;
  red[c] = v * v;
  __syncthreads();
  #pragma unroll
  for (int s = 128; s > 0; s >>= 1) {
    if (c < s) red[c] += red[c + s];
    __syncthreads();
  }
  if (c == 0) {
    const float nrm = sqrtf(red[0]);
    sinv = 1.0f / fmaxf(nrm, 1e-12f);
    invn[n] = sinv;
  }
  __syncthreads();
  znb[(size_t)n * CDIM + c] = __float2bfloat16(v * sinv);
}

// ---------------------------------------------------------------------------
// Kernel 2: normalize codebook -> bf16 en [E_CODES][CDIM] + fp32 s_e = ||en||^2
// ---------------------------------------------------------------------------
__global__ void __launch_bounds__(256)
vq_prep_e(const float* __restrict__ emb,
          __hip_bfloat16* __restrict__ enb,
          float* __restrict__ s_e) {
  const int e = blockIdx.x;          // 0..16383
  const int c = threadIdx.x;
  const float v = emb[(size_t)e * CDIM + c];

  __shared__ float red[256];
  __shared__ float sinv;
  red[c] = v * v;
  __syncthreads();
  #pragma unroll
  for (int s = 128; s > 0; s >>= 1) {
    if (c < s) red[c] += red[c + s];
    __syncthreads();
  }
  if (c == 0) sinv = 1.0f / fmaxf(sqrtf(red[0]), 1e-12f);
  __syncthreads();
  const float nv = v * sinv;
  enb[(size_t)e * CDIM + c] = __float2bfloat16(nv);
  __syncthreads();                   // red reuse
  red[c] = nv * nv;
  __syncthreads();
  #pragma unroll
  for (int s = 128; s > 0; s >>= 1) {
    if (c < s) red[c] += red[c + s];
    __syncthreads();
  }
  if (c == 0) s_e[e] = red[0];
}

// ---------------------------------------------------------------------------
// Kernel 3: similarity GEMM (bf16 WMMA) + per-row argmax over a column chunk.
//   score[n][e] = 2 * dot(zn[n], en[e]) - s_e[e]   (== const - d[n][e])
// grid = (64 row-groups, 4 column chunks), block = 256 (8 waves x 16 rows).
// B tiles double-buffered in LDS; staged by the Tensor Data Mover (wave 0
// issues tensor_load_to_lds, waits TENSORcnt, barrier releases consumers).
// Per-lane running argmax keeps shuffles out of the hot loop.
// ---------------------------------------------------------------------------
__global__ void __launch_bounds__(256)
vq_score_argmax(const unsigned short* __restrict__ znb,
                const unsigned short* __restrict__ enb,
                const float* __restrict__ s_e,
                float* __restrict__ bestv_out,
                int*   __restrict__ besti_out) {
  __shared__ alignas(16) unsigned short tB[2][TILE_COLS * CDIM]; // 2 x 32 KB

  const int tid  = threadIdx.x;
  const int wave = tid >> 5;
  const int lane = tid & 31;
  const int half = lane >> 4;        // 0: lanes 0-15, 1: lanes 16-31
  const int l16  = lane & 15;
  const int rowbase = blockIdx.x * ROWS_PER_WG + wave * 16;
  const int colbase = blockIdx.y * COLS_PER_CHUNK;

  // ---- A: this wave's 16 rows x 256 K kept in registers, WMMA 16-bit A layout
  // lane<16: row=l16, ktile elems {k..k+7, k+16..k+23}; lane>=16: {k+8.., k+24..}
  bf16x16 A[8];
  {
    const unsigned short* rp = znb + (size_t)(rowbase + l16) * CDIM;
    #pragma unroll
    for (int kt = 0; kt < 8; ++kt) {
      ABPack p;
      p.q[0] = *(const u32x4*)(rp + kt * 32 + (half ? 8  : 0));
      p.q[1] = *(const u32x4*)(rp + kt * 32 + (half ? 24 : 16));
      A[kt] = p.v;
    }
  }

  // Per-lane running best (columns seen by this lane are strictly increasing,
  // so strict '>' preserves argmax-first tie-breaking within a lane).
  float bestv[8];
  int   besti[8];
  #pragma unroll
  for (int r = 0; r < 8; ++r) { bestv[r] = -3.0e38f; besti[r] = 0; }

#ifdef VQ_HAVE_TDM
  const unsigned lds0 = (unsigned)(uintptr_t)(&tB[0][0]);
  const unsigned lds1 = (unsigned)(uintptr_t)(&tB[1][0]);
  if (wave == 0) {
    tdm_load_tile(lds0, enb + (size_t)colbase * CDIM);
    __builtin_amdgcn_s_wait_tensorcnt(0);
  }
  __syncthreads();
#else
  // Fallback: global -> regs -> LDS staging
  u32x4 stage[8];
  auto load_tile = [&](int it) {
    const u32x4* src = (const u32x4*)(enb + (size_t)(colbase + it * TILE_COLS) * CDIM);
    #pragma unroll
    for (int j = 0; j < 8; ++j) stage[j] = src[j * 256 + tid];
  };
  auto store_tile = [&](int buf) {
    u32x4* dst = (u32x4*)tB[buf];
    #pragma unroll
    for (int j = 0; j < 8; ++j) dst[j * 256 + tid] = stage[j];
  };
  load_tile(0);
  store_tile(0);
  __syncthreads();
#endif

  for (int it = 0; it < TILES_PER_CHUNK; ++it) {
    const int cur = it & 1;
    const bool more = (it + 1 < TILES_PER_CHUNK);

#ifdef VQ_HAVE_TDM
    if (more && wave == 0)   // DMA next tile into the other buffer (async)
      tdm_load_tile(cur ? lds0 : lds1,
                    enb + (size_t)(colbase + (it + 1) * TILE_COLS) * CDIM);
#else
    if (more) load_tile(it + 1);   // global loads overlap the WMMA block
#endif

    #pragma unroll
    for (int sub = 0; sub < 4; ++sub) {
      f32x8 acc = {};
      // B layout (32x16, 16-bit): lane<16 holds col n=l16, k {0..7, 16..23}; lane>=16 k {8..15, 24..31}
      const unsigned short* bb = tB[cur] + (sub * 16 + l16) * CDIM + (half ? 8 : 0);
      #pragma unroll
      for (int kt = 0; kt < 8; ++kt) {
        ABPack p;
        p.q[0] = *(const u32x4*)(bb + kt * 32);
        p.q[1] = *(const u32x4*)(bb + kt * 32 + 16);
        acc = __builtin_amdgcn_wmma_f32_16x16x32_bf16(false, A[kt], false, p.v,
                                                      (short)0, acc, false, false);
      }
      const int   col = colbase + it * TILE_COLS + sub * 16 + l16;
      const float se  = s_e[col];                 // L2-resident, 64 KB table
      // C layout: lanes 0-15 hold M=r (VGPR r), lanes 16-31 hold M=8+r.
      #pragma unroll
      for (int r = 0; r < 8; ++r) {
        const float sc  = fmaf(2.0f, acc[r], -se);
        const bool  upd = sc > bestv[r];
        bestv[r] = upd ? sc  : bestv[r];
        besti[r] = upd ? col : besti[r];
      }
    }

#ifdef VQ_HAVE_TDM
    if (more && wave == 0) __builtin_amdgcn_s_wait_tensorcnt(0);
#else
    if (more) store_tile((it + 1) & 1);
#endif
    __syncthreads();                      // one barrier per iteration
  }

  // ---- final 16-lane butterfly argmax (once), tie-break on smaller index
  #pragma unroll
  for (int r = 0; r < 8; ++r) {
    float sc = bestv[r];
    int   ci = besti[r];
    #pragma unroll
    for (int m = 1; m < 16; m <<= 1) {
      const float ov = __shfl_xor(sc, m, 32);
      const int   oi = __shfl_xor(ci, m, 32);
      const bool  t  = (ov > sc) || (ov == sc && oi < ci);
      sc = t ? ov : sc;
      ci = t ? oi : ci;
    }
    bestv[r] = sc;
    besti[r] = ci;
  }

  // lane 0 writes rows rowbase+0..7, lane 16 writes rowbase+8..15
  if (l16 == 0) {
    #pragma unroll
    for (int r = 0; r < 8; ++r) {
      const int row = rowbase + half * 8 + r;
      bestv_out[(size_t)row * COL_CHUNKS + blockIdx.y] = bestv[r];
      besti_out[(size_t)row * COL_CHUNKS + blockIdx.y] = besti[r];
    }
  }
}

// ---------------------------------------------------------------------------
// Kernel 4: merge the 4 column-chunk partial argmaxes
// ---------------------------------------------------------------------------
__global__ void __launch_bounds__(256)
vq_merge(const float* __restrict__ bestv, const int* __restrict__ besti,
         int* __restrict__ idx_out) {
  const int row = blockIdx.x * 256 + threadIdx.x;
  if (row >= N_ROWS) return;
  float bv = bestv[row * COL_CHUNKS];
  int   bi = besti[row * COL_CHUNKS];
  #pragma unroll
  for (int c = 1; c < COL_CHUNKS; ++c) {
    const float v = bestv[row * COL_CHUNKS + c];
    const int   i = besti[row * COL_CHUNKS + c];
    const bool  t = (v > bv) || (v == bv && i < bi);
    bv = t ? v : bv;
    bi = t ? i : bi;
  }
  idx_out[row] = bi;
}

// ---------------------------------------------------------------------------
// Kernel 5: gather raw emb[idx] -> z_q (NCHW), idx-as-float, per-row MSE partial
// ---------------------------------------------------------------------------
__global__ void __launch_bounds__(256)
vq_output(const float* __restrict__ z,
          const float* __restrict__ emb,
          const float* __restrict__ invn,
          const int*  __restrict__ idx,
          float* __restrict__ zq_out,
          float* __restrict__ idxf_out,
          float* __restrict__ partial) {
  const int n = blockIdx.x, c = threadIdx.x;
  const int b = n >> 10, hw = n & (HW - 1);
  const int e = idx[n];
  const float q = emb[(size_t)e * CDIM + c];     // RAW embedding, per reference
  const size_t zi = (size_t)b * (CDIM * HW) + (size_t)c * HW + hw;
  zq_out[zi] = q;
  const float znv = z[zi] * invn[n];
  const float d = q - znv;

  __shared__ float red[256];
  red[c] = d * d;
  __syncthreads();
  #pragma unroll
  for (int s = 128; s > 0; s >>= 1) {
    if (c < s) red[c] += red[c + s];
    __syncthreads();
  }
  if (c == 0) {
    partial[n] = red[0];
    idxf_out[n] = (float)e;
  }
}

// ---------------------------------------------------------------------------
// Kernel 6: deterministic final loss reduction (single block)
// ---------------------------------------------------------------------------
__global__ void __launch_bounds__(256)
vq_loss(const float* __restrict__ partial, float* __restrict__ loss_out) {
  __shared__ float red[256];
  const int t = threadIdx.x;
  float s = 0.0f;
  for (int i = t; i < N_ROWS; i += 256) s += partial[i];
  red[t] = s;
  __syncthreads();
  #pragma unroll
  for (int st = 128; st > 0; st >>= 1) {
    if (t < st) red[t] += red[t + st];
    __syncthreads();
  }
  if (t == 0)
    loss_out[0] = 2.25f * red[0] / (float)((size_t)N_ROWS * CDIM);
}

// ---------------------------------------------------------------------------
// Launcher
// ---------------------------------------------------------------------------
extern "C" void kernel_launch(void* const* d_in, const int* in_sizes, int n_in,
                              void* d_out, int out_size, void* d_ws, size_t ws_size,
                              hipStream_t stream) {
  const float* z   = (const float*)d_in[0];   // [8,256,32,32]
  const float* emb = (const float*)d_in[1];   // [16384,256]

  // Workspace layout (bytes)
  char* ws = (char*)d_ws;
  __hip_bfloat16* znb  = (__hip_bfloat16*)(ws + 0);            //  4 MB
  __hip_bfloat16* enb  = (__hip_bfloat16*)(ws + 4194304);      //  8 MB
  float* s_e     = (float*)(ws + 12582912);                    // 64 KB
  float* invn    = (float*)(ws + 12648448);                    // 32 KB
  int*   idx     = (int*)  (ws + 12681216);                    // 32 KB
  float* partial = (float*)(ws + 12713984);                    // 32 KB
  float* bestv   = (float*)(ws + 12746752);                    // 128 KB
  int*   besti   = (int*)  (ws + 12877824);                    // 128 KB

  float* out   = (float*)d_out;
  float* zq    = out;                          // 2,097,152 floats (NCHW)
  float* idxf  = out + 2097152;                // 8,192 floats
  float* lossp = out + 2097152 + 8192;         // 1 float

  vq_prep_z<<<dim3(N_ROWS), dim3(256), 0, stream>>>(z, znb, invn);
  vq_prep_e<<<dim3(E_CODES), dim3(256), 0, stream>>>(emb, enb, s_e);
  vq_score_argmax<<<dim3(N_ROWS / ROWS_PER_WG, COL_CHUNKS), dim3(256), 0, stream>>>(
      (const unsigned short*)znb, (const unsigned short*)enb, s_e, bestv, besti);
  vq_merge<<<dim3(N_ROWS / 256), dim3(256), 0, stream>>>(bestv, besti, idx);
  vq_output<<<dim3(N_ROWS), dim3(256), 0, stream>>>(z, emb, invn, idx, zq, idxf, partial);
  vq_loss<<<dim3(1), dim3(256), 0, stream>>>(partial, lossp);
}
